// STU_86002425135132
// MI455X (gfx1250) — matrix operand order repaired and checked by reading, not compiled, not run on hardware
//
#include <hip/hip_runtime.h>
#include <hip/hip_bf16.h>
#include <stddef.h>
#include <stdint.h>

// ---------------------------------------------------------------------------
// STU layer for MI455X (gfx1250, wave32, WMMA).
//   Stage 0: convert x, M_plus/M_minus (transposed), phi (+alt-sign copy) to bf16
//   Stage 1: w_s[b,t,k,d] = x[b,t,:] @ M_s[k]      (v_wmma_f32_16x16x32_bf16)
//   Stage 2: y[b,t,d] = sum_{s,k,sign} phi_sign[k,s] * w_sign[b,t-s,k,d]
//            causal Toeplitz contraction on WMMA; w blocks staged into LDS via
//            global_load_async_to_lds_b128 and shared by 4 waves (4 t-tiles).
// Intermediates kept in bf16 to halve HBM/L2 traffic (201 MB ~ fits 192MB L2).
// ---------------------------------------------------------------------------

#define Bdim 2
#define Tdim 2048
#define Cdim 768
#define Ldim 2048
#define Kdim 16
#define Ddim 768

typedef __attribute__((ext_vector_type(16))) __bf16 v16bf;
typedef __attribute__((ext_vector_type(8)))  __bf16 v8bf;
typedef __attribute__((ext_vector_type(8), aligned(2))) __bf16 v8bf_u; // unaligned load view
typedef __attribute__((ext_vector_type(8)))  float  v8f;

// ------------------------- stage 0: conversions ----------------------------
__global__ void stu_cvt_x(const float* __restrict__ x, __bf16* __restrict__ xb, int n) {
    int i = blockIdx.x * 256 + threadIdx.x;
    if (i < n) xb[i] = (__bf16)x[i];
}

// Mt[sign][k][d][c] = bf16(M_sign[k][c][d])   (transpose so stage-1 B-frags are contiguous in c)
__global__ void stu_cvt_M(const float* __restrict__ Mp, const float* __restrict__ Mm,
                          __bf16* __restrict__ Mt) {
    size_t i = (size_t)blockIdx.x * 256 + threadIdx.x;
    const size_t total = (size_t)2 * Kdim * Ddim * Cdim;
    if (i >= total) return;
    int c = (int)(i % Cdim); size_t r = i / Cdim;
    int d = (int)(r % Ddim); r /= Ddim;
    int k = (int)(r % Kdim); int sign = (int)(r / Kdim);
    const float* M = sign ? Mm : Mp;
    Mt[i] = (__bf16)M[((size_t)k * Cdim + c) * Ddim + d];
}

// phib[sign][k][s] = bf16( phi[s,k] * (sign ? (-1)^s : 1) )
__global__ void stu_cvt_phi(const float* __restrict__ phi, __bf16* __restrict__ phib) {
    int i = blockIdx.x * 256 + threadIdx.x;
    if (i >= 2 * Kdim * Ldim) return;
    int s = i % Ldim; int r = i / Ldim;
    int k = r % Kdim; int sign = r / Kdim;
    float v = phi[(size_t)s * Kdim + k];
    if (sign && (s & 1)) v = -v;
    phib[i] = (__bf16)v;
}

// ------------------------- stage 1: GEMM w = x @ M -------------------------
// grid = (B*T/16, D/16, 2*K), block = 32 (one wave per 16x16 tile)
// w layout: [sign][b][k][d][t]  (time-contiguous rows for stage-2 B fragments)
__global__ __launch_bounds__(32) void stu_gemm(const __bf16* __restrict__ xb,
                                               const __bf16* __restrict__ Mt,
                                               __bf16* __restrict__ w) {
    const int lane = threadIdx.x;
    const int half = lane >> 4;        // 0 | 1
    const int lm   = lane & 15;        // row-within-tile for A, col for B/C
    const int mt0  = blockIdx.x * 16;  // flat (b*T + t) tile base
    const int d0   = blockIdx.y * 16;
    const int sk   = blockIdx.z;
    const int sign = sk >> 4, k = sk & 15;

    const __bf16* arow = xb + (size_t)(mt0 + lm) * Cdim;
    const __bf16* brow = Mt + (((size_t)sign * Kdim + k) * Ddim + (d0 + lm)) * (size_t)Cdim;

    v8f acc = {};
    for (int c0 = 0; c0 < Cdim; c0 += 32) {
        // A: 16x32 bf16 fragment; lane holds K = {8h..8h+7} and {16+8h..16+8h+7}
        v8bf a0 = *(const v8bf*)(arow + c0 + 8 * half);
        v8bf a1 = *(const v8bf*)(arow + c0 + 16 + 8 * half);
        // B: 32x16 bf16 fragment; lane holds K = {8h..8h+7} and {16+8h..16+8h+7}
        v8bf b0 = *(const v8bf*)(brow + c0 + 8 * half);
        v8bf b1 = *(const v8bf*)(brow + c0 + 16 + 8 * half);
        v16bf a = __builtin_shufflevector(a0, a1, 0,1,2,3,4,5,6,7, 8,9,10,11,12,13,14,15);
        v16bf b = __builtin_shufflevector(b0, b1, 0,1,2,3,4,5,6,7, 8,9,10,11,12,13,14,15);
        acc = __builtin_amdgcn_wmma_f32_16x16x32_bf16(false, a, false, b,
                                                      (short)0, acc, false, false);
    }

    // C/D layout: element v of lane -> (m = v + 8*half, n = lm); rows of acc are
    // contiguous time steps -> one b128 bf16 store per lane.
    const int bidx = mt0 / Tdim;
    const int t0   = mt0 % Tdim;
    __bf16* wrow = w + ((((size_t)sign * Bdim + bidx) * Kdim + k) * Ddim + (d0 + lm)) * (size_t)Tdim;
    v8bf outv;
#pragma unroll
    for (int v = 0; v < 8; ++v) outv[v] = (__bf16)acc[v];
    *(v8bf*)(wrow + t0 + 8 * half) = outv;
}

// ------------------- stage 2: causal Toeplitz contraction ------------------
// grid = (B * T/64, D/64), block = 128 (4 waves; wave w owns t-tile tg*64+w*16).
// For each (sign,k,s0): 4 KB of w ([d0..d0+63] x [s0..s0+31]) is staged into LDS
// with global_load_async_to_lds_b128; every wave runs 4 WMMAs against it.
__global__ __launch_bounds__(128) void stu_conv(const __bf16* __restrict__ phib,
                                                const __bf16* __restrict__ w,
                                                float* __restrict__ y) {
    __shared__ __bf16 Bsh[64 * 32];                 // 4 KB staging buffer

    const int tid  = threadIdx.x;
    const int wave = tid >> 5;
    const int lane = tid & 31;
    const int half = lane >> 4;
    const int lm   = lane & 15;

    const int b    = blockIdx.x >> 5;               // T/64 = 32 tile-groups per batch
    const int tg   = blockIdx.x & 31;
    const int t0   = tg * 64 + wave * 16;           // this wave's output t-tile
    const int smax = tg * 64 + 64;                  // s0 range shared by all 4 waves
    const int d0   = blockIdx.y * 64;

    const int tt = t0 + lm;                         // this lane's output time row

    // staging assignment: thread -> (row r of 64, 32B chunk of 64B row)
    const int srow = tid >> 1;                      // 0..63
    const int scol = (tid & 1) * 16;                // element offset 0 | 16
    const unsigned ldsBase = (unsigned)(uintptr_t)(&Bsh[0]);
    const unsigned ldsOff  = ldsBase + (unsigned)(srow * 32 + scol) * 2;

    v8f acc[4] = {v8f{}, v8f{}, v8f{}, v8f{}};

    for (int sign = 0; sign < 2; ++sign) {
        for (int k = 0; k < Kdim; ++k) {
            const __bf16* phk   = phib + ((size_t)sign * Kdim + k) * Ldim;
            const __bf16* wbase = w + (((size_t)sign * Bdim + b) * Kdim + k) * (size_t)Ddim * Tdim;

            for (int s0 = 0; s0 < smax; s0 += 32) {
                __syncthreads();                    // prior reads of Bsh complete
                // async DMA: 128 threads x 32 B -> 4 KB tile [d0..d0+63][s0..s0+31]
                {
                    uint64_t g = (uint64_t)(uintptr_t)(wbase + (size_t)(d0 + srow) * Tdim + s0 + scol);
                    asm volatile("global_load_async_to_lds_b128 %0, %1, off"
                                 :: "v"(ldsOff), "v"(g) : "memory");
                    asm volatile("global_load_async_to_lds_b128 %0, %1, off offset:16"
                                 :: "v"(ldsOff), "v"(g) : "memory");
                }
                asm volatile("s_wait_asynccnt 0" ::: "memory");
                __syncthreads();                    // staged data visible to all waves

                if (t0 + 16 > s0) {                 // wave-uniform causal bound
                    // Toeplitz A fragment: A[m][K] = phi[t0+m - (s0+K)] (0 if negative)
                    v16bf a;
                    if (t0 - s0 >= 31) {
                        // interior: two reversed 16-byte runs of phi per lane
                        int base = tt - s0 - 8 * half;              // >= 23
                        v8bf c0 = *(const v8bf_u*)(phk + base - 7);
                        v8bf c1 = *(const v8bf_u*)(phk + base - 23);
                        a = __builtin_shufflevector(c0, c1,
                                7,6,5,4,3,2,1,0, 15,14,13,12,11,10,9,8);
                    } else {
                        // causal boundary: masked scalar gather
#pragma unroll
                        for (int e = 0; e < 16; ++e) {
                            int Kk   = (e < 8) ? (8 * half + e) : (16 + 8 * half + (e - 8));
                            int diff = tt - s0 - Kk;
                            __bf16 pv = phk[diff >= 0 ? diff : 0];
                            a[e] = (diff >= 0) ? pv : (__bf16)0.0f;
                        }
                    }
                    // 4 WMMAs share A; B fragments come from LDS (ds_load_b128)
#pragma unroll
                    for (int tile = 0; tile < 4; ++tile) {
                        const __bf16* bp = &Bsh[(tile * 16 + lm) * 32 + 8 * half];
                        v8bf b0 = *(const v8bf*)(bp);
                        v8bf b1 = *(const v8bf*)(bp + 16);
                        v16bf bb = __builtin_shufflevector(b0, b1,
                                       0,1,2,3,4,5,6,7, 8,9,10,11,12,13,14,15);
                        acc[tile] = __builtin_amdgcn_wmma_f32_16x16x32_bf16(
                                        false, a, false, bb, (short)0, acc[tile], false, false);
                    }
                }
            }
        }
    }

    // y[b][t][d] f32: element v -> (t = t0 + v + 8*half, d = d0 + tile*16 + lm)
#pragma unroll
    for (int tile = 0; tile < 4; ++tile) {
#pragma unroll
        for (int v = 0; v < 8; ++v) {
            int t = t0 + v + 8 * half;
            int d = d0 + tile * 16 + lm;
            y[((size_t)b * Tdim + t) * Ddim + d] = acc[tile][v];
        }
    }
}

// ---------------------------------------------------------------------------
extern "C" void kernel_launch(void* const* d_in, const int* in_sizes, int n_in,
                              void* d_out, int out_size, void* d_ws, size_t ws_size,
                              hipStream_t stream) {
    const float* x   = (const float*)d_in[0];
    const float* phi = (const float*)d_in[1];
    const float* Mp  = (const float*)d_in[2];
    const float* Mm  = (const float*)d_in[3];
    float* y = (float*)d_out;

    // workspace carve-up (all 16B aligned)
    char* ws = (char*)d_ws;
    const size_t xb_bytes   = (size_t)Bdim * Tdim * Cdim * 2;             // 6.3 MB
    const size_t Mt_bytes   = (size_t)2 * Kdim * Ddim * Cdim * 2;         // 37.7 MB
    const size_t phib_bytes = (size_t)2 * Kdim * Ldim * 2;                // 128 KB
    __bf16* xb   = (__bf16*)(ws);
    __bf16* Mt   = (__bf16*)(ws + xb_bytes);
    __bf16* phib = (__bf16*)(ws + xb_bytes + Mt_bytes);
    __bf16* w    = (__bf16*)(ws + xb_bytes + Mt_bytes + phib_bytes);      // 201 MB

    const int nx = Bdim * Tdim * Cdim;
    stu_cvt_x<<<(nx + 255) / 256, 256, 0, stream>>>(x, xb, nx);

    const size_t nM = (size_t)2 * Kdim * Ddim * Cdim;
    stu_cvt_M<<<(unsigned)((nM + 255) / 256), 256, 0, stream>>>(Mp, Mm, Mt);

    const int nphi = 2 * Kdim * Ldim;
    stu_cvt_phi<<<(nphi + 255) / 256, 256, 0, stream>>>(phi, phib);

    // stage 1: 256 x 48 x 32 tiles, one wave each
    stu_gemm<<<dim3(Bdim * Tdim / 16, Ddim / 16, 2 * Kdim), 32, 0, stream>>>(xb, Mt, w);

    // stage 2: (B * T/64) x (D/64) blocks of 4 waves
    stu_conv<<<dim3(Bdim * (Tdim / 64), Ddim / 64), 128, 0, stream>>>(phib, w, y);
}
